// FP8LinearBase_1915555414610
// MI455X (gfx1250) — matrix operand order repaired and checked by reading, not compiled, not run on hardware
//
#include <hip/hip_runtime.h>

typedef __attribute__((ext_vector_type(16))) __bf16 v16bf;
typedef __attribute__((ext_vector_type(8)))  float  v8f;

#define TM 256
#define TN 128
#define TK 64
#define LDSU 36                    // uints per 64-bf16 row: 32 data + 4 pad (row = 144B)
#define ATILEU (256 * LDSU)        // x tile: 256 rows
#define BTILEU (128 * LDSU)        // w tile: 128 rows
#define ABYTES (ATILEU * 4)        // 36864 B
#define BBYTES (BTILEU * 4)        // 18432 B

__device__ __forceinline__ unsigned pack_bf16x2(float a, float b) {
  // round-to-nearest-even fp32 -> bf16, packed pair
  unsigned ua = __float_as_uint(a), ub = __float_as_uint(b);
  ua += 0x7FFFu + ((ua >> 16) & 1u);
  ub += 0x7FFFu + ((ub >> 16) & 1u);
  return (ua >> 16) | (ub & 0xFFFF0000u);
}

// ---------------- Pre-pass 1: x fp32 -> bf16 (bandwidth bound) ----------------
__global__ __launch_bounds__(256) void cvt_x_kernel(const float* __restrict__ x,
                                                    unsigned* __restrict__ xh2,
                                                    int K) {
  const int row = blockIdx.x;
  const int t = threadIdx.x;
  const long long base4 = (long long)row * (K >> 2);
  for (int j = t; j < (K >> 2); j += 256) {
    float4 v = ((const float4*)x)[base4 + j];
    ((uint2*)xh2)[base4 + j] = make_uint2(pack_bf16x2(v.x, v.y), pack_bf16x2(v.z, v.w));
  }
}

// ---------- Pre-pass 2: w fp32 * blockscale -> bf16 (fused dequant) ----------
__global__ __launch_bounds__(256) void cvt_w_kernel(const float* __restrict__ w,
                                                    const float* __restrict__ wsi,
                                                    unsigned* __restrict__ wh2,
                                                    int K) {
  const int row = blockIdx.x;               // n
  const int t = threadIdx.x;
  const int kblocks = K >> 7;
  const int srow = (row >> 7) * kblocks;
  const long long base4 = (long long)row * (K >> 2);
  for (int j = t; j < (K >> 2); j += 256) {
    const float s = wsi[srow + (j >> 5)];   // k = j*4 ; k>>7 == j>>5
    float4 v = ((const float4*)w)[base4 + j];
    ((uint2*)wh2)[base4 + j] =
        make_uint2(pack_bf16x2(v.x * s, v.y * s), pack_bf16x2(v.z * s, v.w * s));
  }
}

// ---------------- GEMM: bf16 WMMA, async-to-LDS double buffered ----------------
union FragAB { v16bf v; uint4 u[2]; };
union FragC  { v8f  v; float f[8]; };

__device__ __forceinline__ void async_b128(unsigned lds_addr, unsigned voff,
                                           unsigned long long sbase) {
  // GVS mode: mem = SADDR(64b) + VADDR(32b offset); VDST = LDS byte address
  asm volatile("global_load_async_to_lds_b128 %0, %1, %2"
               :: "v"(lds_addr), "v"(voff), "s"(sbase) : "memory");
}

__global__ __launch_bounds__(256) void gemm_bf16_wmma(
    const unsigned* __restrict__ xh2,   // bf16 pairs, M x K
    const unsigned* __restrict__ wh2,   // bf16 pairs (dequantized), N x K
    float* __restrict__ y,
    int M, int N, int K) {
  __shared__ unsigned As[2 * ATILEU];   // double-buffered x tile (256 x 64 bf16)
  __shared__ unsigned Bs[2 * BTILEU];   // double-buffered w tile (128 x 64 bf16)

  const int t     = threadIdx.x;
  const int wid   = t >> 5;             // 8 waves (wave32)
  const int lane  = t & 31;
  const int half  = lane >> 4;
  const int r     = lane & 15;
  const int waveM = wid & 3;            // 0..3 -> 64-row stripe
  const int waveN = wid >> 2;           // 0..1 -> 64-col stripe

  const int m0 = blockIdx.y * TM;
  const int n0 = blockIdx.x * TN;

  const unsigned long long xbase = (unsigned long long)xh2 + (unsigned long long)m0 * K * 2ull;
  const unsigned long long wbase = (unsigned long long)wh2 + (unsigned long long)n0 * K * 2ull;

  // Staging: A = 2048 x 16B segs (8/thread), B = 1024 x 16B segs (4/thread)
  unsigned xoffA[8], ldsA[8], xoffB[4], ldsB[4];
  const unsigned asb = (unsigned)(unsigned long long)As;   // LDS byte address of As
  const unsigned bsb = (unsigned)(unsigned long long)Bs;
  #pragma unroll
  for (int it = 0; it < 8; ++it) {
    int id = it * 256 + t, row = id >> 3, cg = id & 7;     // 8 segs per 128B row
    xoffA[it] = (unsigned)(row * K * 2 + cg * 16);
    ldsA[it]  = asb + (unsigned)(row * 144 + cg * 16);
  }
  #pragma unroll
  for (int it = 0; it < 4; ++it) {
    int id = it * 256 + t, row = id >> 3, cg = id & 7;
    xoffB[it] = (unsigned)(row * K * 2 + cg * 16);
    ldsB[it]  = bsb + (unsigned)(row * 144 + cg * 16);
  }

  FragC acc[4][4];
  #pragma unroll
  for (int mi = 0; mi < 4; ++mi)
    #pragma unroll
    for (int ni = 0; ni < 4; ++ni)
      #pragma unroll
      for (int i = 0; i < 8; ++i) acc[mi][ni].f[i] = 0.0f;

  const int nkc = K / TK;               // 64 chunks

  // Preload chunk 0 -> buffer 0
  #pragma unroll
  for (int it = 0; it < 8; ++it) async_b128(ldsA[it], xoffA[it], xbase);
  #pragma unroll
  for (int it = 0; it < 4; ++it) async_b128(ldsB[it], xoffB[it], wbase);

  for (int kc = 0; kc < nkc; ++kc) {
    const int cur = kc & 1;
    asm volatile("s_wait_asynccnt 0x0" ::: "memory");  // my slice of chunk kc is in LDS
    __syncthreads();                                   // everyone's slice is in LDS

    if (kc + 1 < nkc) {                                // prefetch next chunk -> other buffer
      const unsigned kb  = (unsigned)(kc + 1) * 128u;  // TK*2 bytes
      const unsigned loA = (unsigned)((cur ^ 1) * ABYTES);
      const unsigned loB = (unsigned)((cur ^ 1) * BBYTES);
      #pragma unroll
      for (int it = 0; it < 8; ++it) async_b128(ldsA[it] + loA, xoffA[it] + kb, xbase);
      #pragma unroll
      for (int it = 0; it < 4; ++it) async_b128(ldsB[it] + loB, xoffB[it] + kb, wbase);
    }

    const unsigned ca = (unsigned)cur * ATILEU;
    const unsigned cb = (unsigned)cur * BTILEU;
    #pragma unroll
    for (int ks = 0; ks < TK; ks += 32) {
      FragAB a[4], b[4];
      #pragma unroll
      for (int mi = 0; mi < 4; ++mi) {
        // A 16x32 bf16: half 0 -> K {0..7,16..23}, half 1 -> K {8..15,24..31}
        int row = waveM * 64 + mi * 16 + r;
        const unsigned* base = &As[ca + row * LDSU + (ks >> 1) + half * 4];
        a[mi].u[0] = *(const uint4*)(base);
        a[mi].u[1] = *(const uint4*)(base + 8);
      }
      #pragma unroll
      for (int ni = 0; ni < 4; ++ni) {
        // B 32x16 bf16: lane holds column n=r; half 0 -> K {0..7,16..23}, half 1 -> K {8..15,24..31}
        int col = waveN * 64 + ni * 16 + r;
        const unsigned* base = &Bs[cb + col * LDSU + (ks >> 1) + half * 4];
        b[ni].u[0] = *(const uint4*)(base);
        b[ni].u[1] = *(const uint4*)(base + 8);
      }
      #pragma unroll
      for (int mi = 0; mi < 4; ++mi)
        #pragma unroll
        for (int ni = 0; ni < 4; ++ni)
          acc[mi][ni].v = __builtin_amdgcn_wmma_f32_16x16x32_bf16(
              false, a[mi].v, false, b[ni].v, (short)0, acc[mi][ni].v, false, false);
    }
    __syncthreads();                                   // release current buffer
  }

  // Epilogue: C/D layout -> VGPR i is row half*8+i, column = lane r
  #pragma unroll
  for (int mi = 0; mi < 4; ++mi)
    #pragma unroll
    for (int ni = 0; ni < 4; ++ni) {
      int col = n0 + waveN * 64 + ni * 16 + r;
      #pragma unroll
      for (int i = 0; i < 8; ++i) {
        int rowg = m0 + waveM * 64 + mi * 16 + half * 8 + i;
        y[(size_t)rowg * N + col] = acc[mi][ni].f[i];
      }
    }
}

extern "C" void kernel_launch(void* const* d_in, const int* in_sizes, int n_in,
                              void* d_out, int out_size, void* d_ws, size_t ws_size,
                              hipStream_t stream) {
  const float* x   = (const float*)d_in[0];
  const float* w   = (const float*)d_in[1];
  const float* wsi = (const float*)d_in[2];
  float* y = (float*)d_out;

  const int K = 4096;
  const int M = in_sizes[0] / K;   // 8192
  const int N = in_sizes[1] / K;   // 4096

  // Workspace layout (bf16 copies): xh = M*K*2 B, wh = N*K*2 B  (~100 MB total)
  unsigned* xh2 = (unsigned*)d_ws;
  unsigned* wh2 = xh2 + ((size_t)M * K / 2);

  cvt_x_kernel<<<dim3(M), 256, 0, stream>>>(x, xh2, K);
  cvt_w_kernel<<<dim3(N), 256, 0, stream>>>(w, wsi, wh2, K);

  dim3 grid(N / TN, M / TM);       // 32 x 32 workgroups
  gemm_bf16_wmma<<<grid, 256, 0, stream>>>(xh2, wh2, y, M, N, K);
}
